// PD_Block_68332929679630
// MI455X (gfx1250) — matrix operand order repaired and checked by reading, not compiled, not run on hardware
//
#include <hip/hip_runtime.h>
#include <math.h>

typedef __attribute__((ext_vector_type(16))) _Float16 v16h;
typedef __attribute__((ext_vector_type(8)))  float    v8f;

#define T_TOK 2048
#define B_SZ  4
#define L_SEQ 512
#define E_DIM 128
#define N_DIM 128
#define H_DIM 256
#define K_SEL 8
#define TWO_PI_F 6.283185307179586f

#define ACT_NONE 0
#define ACT_GELU 1
#define ACT_SIG  2

__device__ __forceinline__ float fast_rcp(float x) {
    return __builtin_amdgcn_rcpf(x);
}

template <int ACT>
__device__ __forceinline__ float apply_act(float r) {
    if (ACT == ACT_GELU) return 0.5f * r * (1.0f + erff(r * 0.70710678118654752f));
    if (ACT == ACT_SIG)  return fast_rcp(1.0f + __expf(-r));
    return r;
}

// ---------------------------------------------------------------------------
// WMMA GEMM: Y[T_TOK x NN] = act( X[T_TOK x KD] * W[NN x KD]^T + bias )
// Each wave computes a 16(M) x 64(N) strip: A fragment loaded once per
// K-chunk and reused by 4 independent v_wmma accumulators (hides XDL latency,
// no RAW hazards). K-loop fully unrolled via template. 4 waves per block.
// ---------------------------------------------------------------------------
template <int KD, int NN, int ACT>
__global__ void pd_wmma_gemm_kernel(const float* __restrict__ X,
                                    const float* __restrict__ W,
                                    const float* __restrict__ bias,
                                    float* __restrict__ Y) {
    constexpr int STRIPS = NN / 64;                 // 16x64 strips per row-tile
    const int wave = threadIdx.x >> 5;
    const int tile = blockIdx.x * 4 + wave;         // total waves sized exactly
    const int t0 = (tile / STRIPS) << 4;            // token-tile origin (M)
    const int nb = (tile % STRIPS) * 64;            // strip origin (N)

    const int lane = threadIdx.x & 31;
    const int hi   = lane >> 4;                     // K-half selector
    const int lm   = lane & 15;                     // row (A) / column (B,C)

    const float* xrow = X + (size_t)(t0 + lm) * KD; // A row for this lane
    const float* wr0  = W + (size_t)(nb + lm) * KD; // B columns (W rows), 4 subtiles
    const float* wr1  = wr0 + (size_t)16 * KD;
    const float* wr2  = wr0 + (size_t)32 * KD;
    const float* wr3  = wr0 + (size_t)48 * KD;

    v8f acc0 = {0.f,0.f,0.f,0.f,0.f,0.f,0.f,0.f};
    v8f acc1 = acc0, acc2 = acc0, acc3 = acc0;

#pragma unroll
    for (int k0 = 0; k0 < KD; k0 += 32) {
        v16h a;
        // A 16x32 f16: elems 0..7 -> K=k0+hi*8+j ; elems 8..15 -> K=k0+16+hi*8+j
        const float* pa0 = xrow + k0 + hi * 8;
        const float* pa1 = pa0 + 16;
#pragma unroll
        for (int j = 0; j < 8; ++j) {
            a[j]     = (_Float16)pa0[j];
            a[j + 8] = (_Float16)pa1[j];
        }
        // B 32x16 f16: elems 0..7 -> K=k0+hi*8+j ; elems 8..15 -> K=k0+16+hi*8+j, column = lm
        v16h b0, b1, b2, b3;
#pragma unroll
        for (int j = 0; j < 16; ++j) {
            b0[j] = (_Float16)wr0[k0 + hi * 8 + (j & 7) + ((j >> 3) << 4)];
            b1[j] = (_Float16)wr1[k0 + hi * 8 + (j & 7) + ((j >> 3) << 4)];
            b2[j] = (_Float16)wr2[k0 + hi * 8 + (j & 7) + ((j >> 3) << 4)];
            b3[j] = (_Float16)wr3[k0 + hi * 8 + (j & 7) + ((j >> 3) << 4)];
        }
        acc0 = __builtin_amdgcn_wmma_f32_16x16x32_f16(false, a, false, b0, (short)0, acc0, false, false);
        acc1 = __builtin_amdgcn_wmma_f32_16x16x32_f16(false, a, false, b1, (short)0, acc1, false, false);
        acc2 = __builtin_amdgcn_wmma_f32_16x16x32_f16(false, a, false, b2, (short)0, acc2, false, false);
        acc3 = __builtin_amdgcn_wmma_f32_16x16x32_f16(false, a, false, b3, (short)0, acc3, false, false);
    }

    const v8f* accs[4] = { &acc0, &acc1, &acc2, &acc3 };
#pragma unroll
    for (int s = 0; s < 4; ++s) {
        const int col = nb + s * 16 + lm;
        const float bv = bias ? bias[col] : 0.0f;
        const v8f& a = *accs[s];
#pragma unroll
        for (int v = 0; v < 8; ++v) {
            // C/D 16x16 f32: lane(hi,lm), VGPR v -> row = hi*8 + v
            Y[(size_t)(t0 + hi * 8 + v) * NN + col] = apply_act<ACT>(a[v] + bv);
        }
    }
}

// ---------------------------------------------------------------------------
// sel = softmax(x @ S_w^T) over K=8, one thread per token
// ---------------------------------------------------------------------------
__global__ void pd_sel_softmax_kernel(const float* __restrict__ x,
                                      const float* __restrict__ S_w,
                                      float* __restrict__ sel) {
    int t = blockIdx.x * blockDim.x + threadIdx.x;
    if (t >= T_TOK) return;
    const float* xr = x + (size_t)t * E_DIM;
    float lg[K_SEL];
    float mx = -1e30f;
#pragma unroll
    for (int k = 0; k < K_SEL; ++k) {
        const float* wr = S_w + k * E_DIM;
        float s = 0.f;
        for (int e = 0; e < E_DIM; ++e) s += xr[e] * wr[e];
        lg[k] = s;
        mx = fmaxf(mx, s);
    }
    float den = 0.f;
#pragma unroll
    for (int k = 0; k < K_SEL; ++k) { lg[k] = __expf(lg[k] - mx); den += lg[k]; }
    float inv = fast_rcp(den);
#pragma unroll
    for (int k = 0; k < K_SEL; ++k) sel[(size_t)t * K_SEL + k] = lg[k] * inv;
}

// ---------------------------------------------------------------------------
// idx[t][n] = argmax_m sum_k A_dict[n][m][k] * sel[t][k]
// One block per token, thread = n. A_dict is L2-resident (512 KB).
// ---------------------------------------------------------------------------
__global__ void pd_argmax_kernel(const float* __restrict__ A_dict,
                                 const float* __restrict__ sel,
                                 int* __restrict__ idx) {
    int t = blockIdx.x;
    int n = threadIdx.x;
    const float4* sp = (const float4*)(sel + (size_t)t * K_SEL);
    float4 s0 = sp[0], s1 = sp[1];
    const float* ar = A_dict + (size_t)n * N_DIM * K_SEL;
    float best = -1e30f;
    int bi = 0;
    for (int m = 0; m < N_DIM; ++m) {
        const float4* p = (const float4*)(ar + m * K_SEL);
        float4 q0 = p[0], q1 = p[1];
        float v = q0.x * s0.x + q0.y * s0.y + q0.z * s0.z + q0.w * s0.w +
                  q1.x * s1.x + q1.y * s1.y + q1.z * s1.z + q1.w * s1.w;
        if (v > best) { best = v; bi = m; }
    }
    idx[(size_t)t * N_DIM + n] = bi;
}

// ---------------------------------------------------------------------------
// Dc = mag * exp(i * 2pi * sigmoid_phase)
// ---------------------------------------------------------------------------
__global__ void pd_dphase_kernel(const float* __restrict__ mag,
                                 const float* __restrict__ ph,
                                 float* __restrict__ Dre,
                                 float* __restrict__ Dim) {
    int i = blockIdx.x * blockDim.x + threadIdx.x;
    if (i >= T_TOK * N_DIM) return;
    float m = mag[i];
    float p = TWO_PI_F * ph[i];
    float s, c;
    sincosf(p, &s, &c);
    Dre[i] = m * c;
    Dim[i] = m * s;
}

// ---------------------------------------------------------------------------
// Sequential recurrence: h_t[n] = Dc[n]*h_{t-1}[idx[n]] + b[n], h_0 = e_0
// T_t has one-hot rows -> O(N) scaled LDS gather per step.
// One block per batch element; writes feat = [Re(h) | Im(h)].
// ---------------------------------------------------------------------------
__global__ void pd_scan_kernel(const int* __restrict__ idx,
                               const float* __restrict__ Dre,
                               const float* __restrict__ Dim,
                               const float* __restrict__ br,
                               const float* __restrict__ bi,
                               float* __restrict__ feat) {
    __shared__ float hre[N_DIM];
    __shared__ float him[N_DIM];
    int b = blockIdx.x;
    int n = threadIdx.x;
    hre[n] = (n == 0) ? 1.0f : 0.0f;
    him[n] = 0.0f;
    __syncthreads();
    for (int t = 0; t < L_SEQ; ++t) {
        size_t base = ((size_t)b * L_SEQ + t) * N_DIM;
        int j = idx[base + n];
        float dr = Dre[base + n], di = Dim[base + n];
        float hr = hre[j], hm = him[j];
        __syncthreads();                       // all gathers done before overwrite
        float nr = dr * hr - di * hm + br[base + n];
        float ni = dr * hm + di * hr + bi[base + n];
        hre[n] = nr;
        him[n] = ni;
        __syncthreads();
        size_t fb = ((size_t)b * L_SEQ + t) * (2 * N_DIM);
        feat[fb + n]         = nr;
        feat[fb + N_DIM + n] = ni;
    }
}

// ---------------------------------------------------------------------------
// out = LayerNorm(out_pre + D_skip * x) ; one block per token
// ---------------------------------------------------------------------------
__global__ void pd_final_ln_kernel(const float* __restrict__ out_pre,
                                   const float* __restrict__ x,
                                   const float* __restrict__ D_skip,
                                   const float* __restrict__ nw,
                                   const float* __restrict__ nb,
                                   float* __restrict__ out) {
    __shared__ float red[E_DIM];
    int t = blockIdx.x;
    int e = threadIdx.x;
    float v = out_pre[(size_t)t * E_DIM + e] + D_skip[e] * x[(size_t)t * E_DIM + e];
    red[e] = v;
    __syncthreads();
    for (int s = 64; s > 0; s >>= 1) {
        if (e < s) red[e] += red[e + s];
        __syncthreads();
    }
    float mean = red[0] * (1.0f / 128.0f);
    __syncthreads();
    float d = v - mean;
    red[e] = d * d;
    __syncthreads();
    for (int s = 64; s > 0; s >>= 1) {
        if (e < s) red[e] += red[e + s];
        __syncthreads();
    }
    float var = red[0] * (1.0f / 128.0f);
    out[(size_t)t * E_DIM + e] = d * rsqrtf(var + 1e-5f) * nw[e] + nb[e];
}

// ---------------------------------------------------------------------------
template <int KD, int NN, int ACT>
static inline void launch_gemm(const float* X, const float* W, const float* bias,
                               float* Y, hipStream_t stream) {
    constexpr int numWaves = (T_TOK / 16) * (NN / 64);  // one 16x64 strip per wave
    constexpr int blocks = numWaves / 4;                // 4 waves per block
    pd_wmma_gemm_kernel<KD, NN, ACT><<<blocks, 128, 0, stream>>>(X, W, bias, Y);
}

extern "C" void kernel_launch(void* const* d_in, const int* in_sizes, int n_in,
                              void* d_out, int out_size, void* d_ws, size_t ws_size,
                              hipStream_t stream) {
    (void)in_sizes; (void)n_in; (void)out_size; (void)ws_size;
    const float* x      = (const float*)d_in[0];
    const float* norm_w = (const float*)d_in[1];
    const float* norm_b = (const float*)d_in[2];
    const float* mag_w1 = (const float*)d_in[3];
    const float* mag_b1 = (const float*)d_in[4];
    const float* mag_w2 = (const float*)d_in[5];
    const float* mag_b2 = (const float*)d_in[6];
    const float* ph_w1  = (const float*)d_in[7];
    const float* ph_b1  = (const float*)d_in[8];
    const float* ph_w2  = (const float*)d_in[9];
    const float* ph_b2  = (const float*)d_in[10];
    const float* S_w    = (const float*)d_in[11];
    const float* A_dict = (const float*)d_in[12];
    const float* B_re   = (const float*)d_in[13];
    const float* B_im   = (const float*)d_in[14];
    const float* D_skip = (const float*)d_in[15];
    const float* ro_w   = (const float*)d_in[16];
    const float* ro_b   = (const float*)d_in[17];
    float* out = (float*)d_out;

    // Workspace carve-up (floats)
    float* ws = (float*)d_ws;
    size_t off = 0;
    float* sel     = ws + off; off += (size_t)T_TOK * K_SEL;
    float* h1      = ws + off; off += (size_t)T_TOK * H_DIM;   // reused by mag & ph MLPs
    float* mag     = ws + off; off += (size_t)T_TOK * N_DIM;
    float* phs     = ws + off; off += (size_t)T_TOK * N_DIM;
    float* Dre     = ws + off; off += (size_t)T_TOK * N_DIM;
    float* Dim     = ws + off; off += (size_t)T_TOK * N_DIM;
    float* br      = ws + off; off += (size_t)T_TOK * N_DIM;
    float* bi      = ws + off; off += (size_t)T_TOK * N_DIM;
    float* feat    = ws + off; off += (size_t)T_TOK * 2 * N_DIM;
    float* out_pre = ws + off; off += (size_t)T_TOK * E_DIM;
    int*   idx     = (int*)(ws + off);

    // 1) selection softmax
    pd_sel_softmax_kernel<<<(T_TOK + 127) / 128, 128, 0, stream>>>(x, S_w, sel);

    // 2) magnitude MLP (WMMA): x -> h1 (GELU) -> mag (sigmoid)
    launch_gemm<E_DIM, H_DIM, ACT_GELU>(x,  mag_w1, mag_b1, h1,  stream);
    launch_gemm<H_DIM, N_DIM, ACT_SIG >(h1, mag_w2, mag_b2, mag, stream);
    // 3) phase MLP (WMMA); h1 safely reused (stream order)
    launch_gemm<E_DIM, H_DIM, ACT_GELU>(x,  ph_w1,  ph_b1,  h1,  stream);
    launch_gemm<H_DIM, N_DIM, ACT_SIG >(h1, ph_w2,  ph_b2,  phs, stream);
    // 4) complex input projection (WMMA)
    launch_gemm<E_DIM, N_DIM, ACT_NONE>(x, B_re, nullptr, br, stream);
    launch_gemm<E_DIM, N_DIM, ACT_NONE>(x, B_im, nullptr, bi, stream);

    // 5) Dc = mag * exp(i*2pi*phs)
    pd_dphase_kernel<<<(T_TOK * N_DIM + 255) / 256, 256, 0, stream>>>(mag, phs, Dre, Dim);

    // 6) per-row argmax of M = A_dict . sel  (one-hot structure of P)
    pd_argmax_kernel<<<T_TOK, N_DIM, 0, stream>>>(A_dict, sel, idx);

    // 7) sequential scan: scaled-gather recurrence, 4 independent chains
    pd_scan_kernel<<<B_SZ, N_DIM, 0, stream>>>(idx, Dre, Dim, br, bi, feat);

    // 8) readout GEMM (WMMA) + skip/LayerNorm epilogue
    launch_gemm<2 * N_DIM, E_DIM, ACT_NONE>(feat, ro_w, ro_b, out_pre, stream);
    pd_final_ln_kernel<<<T_TOK, E_DIM, 0, stream>>>(out_pre, x, D_skip, norm_w, norm_b, out);
}
